// InstanceNet_28896539967498
// MI455X (gfx1250) — matrix operand order repaired and checked by reading, not compiled, not run on hardware
//
#include <hip/hip_runtime.h>

typedef __attribute__((ext_vector_type(16))) __bf16        v16bf;
typedef __attribute__((ext_vector_type(8)))  float         v8f;
typedef __attribute__((ext_vector_type(8)))  unsigned int  v8u;

constexpr int Bsz = 512, Nsz = 512, Fsz = 512, Dsz = 256, Ksel = 10;
constexpr int ROWS_PER_WG   = 128;                 // 8 waves x 16 rows
constexpr int LDS_STRIDE_U  = 132;                 // 264 bf16 per row (8-bf16 pad) in uints

__device__ __forceinline__ unsigned int pack_bf16(float lo, float hi) {
    unsigned int ul = __float_as_uint(lo);
    unsigned int uh = __float_as_uint(hi);
    ul = (ul + 0x7FFFu + ((ul >> 16) & 1u)) >> 16;           // round-to-nearest-even
    uh = (uh + 0x7FFFu + ((uh >> 16) & 1u)) >> 16;
    return (uh << 16) | (ul & 0xFFFFu);
}

// ---------------- Kernel 0: W f32 -> packed bf16 (row-major, stride 128 uints) -------------
__global__ void convert_w_kernel(const float* __restrict__ W, unsigned int* __restrict__ Wb) {
    int i = blockIdx.x * blockDim.x + threadIdx.x;           // 32768 packed pairs
    if (i < (Dsz * Dsz) / 2) Wb[i] = pack_bf16(W[2 * i], W[2 * i + 1]);
}

// ---------------- Kernel 1: fused bilinear  pred = attn * (dis . (W drug)) -----------------
__global__ __launch_bounds__(256) void bilinear_wmma_kernel(
        const float* __restrict__ ins,            // (B,N,F) f32
        const float* __restrict__ attn,           // (B,N,1) f32
        const unsigned int* __restrict__ Wb,      // (256,128) packed bf16
        float* __restrict__ pred) {               // (B*N) f32
    __shared__ unsigned int Wlds[256 * LDS_STRIDE_U];        // 132 KB (<= 320 KB WGP LDS)

    const int t = threadIdx.x;

    // Stage bf16 W into LDS with the CDNA5 async DMA path (ASYNCcnt-tracked, no VGPR
    // round-trip), so the L2->LDS transfer overlaps the A-tile HBM fetch below.
    for (int i = t; i < 256 * 32; i += 256) {                // 8192 x b128 transfers
        int row = i >> 5, c4 = i & 31;
        unsigned long long gaddr =
            (unsigned long long)(const void*)(Wb + row * 128 + c4 * 4);
        unsigned int ldsOff =
            (unsigned int)(uintptr_t)(&Wlds[row * LDS_STRIDE_U + c4 * 4]);
        asm volatile("global_load_async_to_lds_b128 %0, %1, off"
                     :: "v"(ldsOff), "v"(gaddr) : "memory");
    }

    const int lane = t & 31;
    const int wave = t >> 5;
    const int half = lane >> 4;                               // K-interleave half
    const int m    = lane & 15;                               // row (A) / col (B) within tile
    const long rowBase = (long)blockIdx.x * ROWS_PER_WG + wave * 16;   // flat b*N+n

    // Build 16x256 bf16 A tile (drug) in registers while the async staging runs.
    // 16-bit A layout: lanes 0-15 hold K = kb*32 + [0..7],[16..23]; lanes 16-31 get +8.
    v16bf A[8];
    const float* drow = ins + (size_t)(rowBase + m) * Fsz;
    #pragma unroll
    for (int kb = 0; kb < 8; ++kb) {
        const float* p0 = drow + kb * 32 + half * 8;
        const float* p1 = p0 + 16;
        v8u pk;
        #pragma unroll
        for (int j = 0; j < 4; ++j) pk[j]     = pack_bf16(p0[2 * j], p0[2 * j + 1]);
        #pragma unroll
        for (int j = 0; j < 4; ++j) pk[4 + j] = pack_bf16(p1[2 * j], p1[2 * j + 1]);
        A[kb] = __builtin_bit_cast(v16bf, pk);
    }

    asm volatile("s_wait_asynccnt 0x0" ::: "memory");         // this wave's DMA done
    __syncthreads();                                          // all waves' staging visible

    float rowsum[8];
    #pragma unroll
    for (int j = 0; j < 8; ++j) rowsum[j] = 0.0f;

    #pragma unroll 1
    for (int eb = 0; eb < 16; ++eb) {
        v8f c = {};
        #pragma unroll
        for (int kb = 0; kb < 8; ++kb) {
            // B = W^T: lane's column e = eb*16+m is row e of W,
            // same K layout as A (lanes 0-15: K=kb*32+[0..7],[16..23]; lanes 16-31: +8).
            const unsigned int* bp = &Wlds[(eb * 16 + m) * LDS_STRIDE_U + kb * 16 + half * 4];
            v8u bu;
            #pragma unroll
            for (int j = 0; j < 8; ++j) bu[j] = bp[j < 4 ? j : j + 4];
            v16bf Bm = __builtin_bit_cast(v16bf, bu);
            c = __builtin_amdgcn_wmma_f32_16x16x32_bf16(false, A[kb], false, Bm,
                                                        (short)0, c, false, false);
        }
        // Fuse dis dot-product in C layout: VGPR j -> row rowBase + j + half*8, col = lane&15.
        #pragma unroll
        for (int j = 0; j < 8; ++j) {
            size_t r = (size_t)(rowBase + j + half * 8);
            float dv = ins[r * Fsz + Dsz + eb * 16 + m];
            rowsum[j] = fmaf(c[j], dv, rowsum[j]);
        }
    }

    // Reduce the 16 column-lanes of each half; lane 0 -> rows 0..7, lane 16 -> rows 8..15.
    #pragma unroll
    for (int j = 0; j < 8; ++j) {
        float v = rowsum[j];
        #pragma unroll
        for (int mask = 1; mask < 16; mask <<= 1) v += __shfl_xor(v, mask, 32);
        rowsum[j] = v;
    }
    if (m == 0) {
        #pragma unroll
        for (int j = 0; j < 8; ++j) {
            long r = rowBase + half * 8 + j;
            pred[r] = rowsum[j] * attn[r];
        }
    }
}

// ---------------- Kernel 2: per-batch top-10 mean ------------------------------------------
__global__ __launch_bounds__(256) void topk_mean_kernel(const float* __restrict__ pred,
                                                        float* __restrict__ out) {
    __shared__ float vals[512];
    __shared__ float smax[256];
    __shared__ int   sidx[256];
    const int b = blockIdx.x, t = threadIdx.x;
    vals[t]       = pred[(size_t)b * Nsz + t];
    vals[t + 256] = pred[(size_t)b * Nsz + t + 256];
    __syncthreads();
    float total = 0.0f;
    for (int k = 0; k < Ksel; ++k) {
        float m0 = vals[t], m1 = vals[t + 256];
        smax[t] = fmaxf(m0, m1);
        sidx[t] = (m1 > m0) ? (t + 256) : t;
        __syncthreads();
        for (int s = 128; s > 0; s >>= 1) {
            if (t < s && smax[t + s] > smax[t]) { smax[t] = smax[t + s]; sidx[t] = sidx[t + s]; }
            __syncthreads();
        }
        if (t == 0) { total += smax[0]; vals[sidx[0]] = -__builtin_inff(); }
        __syncthreads();
    }
    if (t == 0) out[b] = total * (1.0f / Ksel);
}

extern "C" void kernel_launch(void* const* d_in, const int* in_sizes, int n_in,
                              void* d_out, int out_size, void* d_ws, size_t ws_size,
                              hipStream_t stream) {
    const float* ins  = (const float*)d_in[0];   // (512,512,512)
    const float* attn = (const float*)d_in[1];   // (512,512,1)
    const float* W    = (const float*)d_in[2];   // (256,256)
    float* out = (float*)d_out;                  // (512,1)

    float*        pred = (float*)d_ws;                                        // 1 MB
    unsigned int* Wb   = (unsigned int*)((char*)d_ws +
                                         (size_t)Bsz * Nsz * sizeof(float));  // +128 KB

    convert_w_kernel<<<(Dsz * Dsz / 2 + 255) / 256, 256, 0, stream>>>(W, Wb);
    bilinear_wmma_kernel<<<(Bsz * Nsz) / ROWS_PER_WG, 256, 0, stream>>>(ins, attn, Wb, pred);
    topk_mean_kernel<<<Bsz, 256, 0, stream>>>(pred, out);
}
